// WindowRegionDetector_52733608460405
// MI455X (gfx1250) — matrix operand order repaired and checked by reading, not compiled, not run on hardware
//
#include <hip/hip_runtime.h>
#include <hip/hip_bf16.h>

// Fused WindowRegionDetector for MI455X (gfx1250, wave32).
// One workgroup = one 128x128 output tile; ~364 MB total DRAM traffic
// (1.56x halo redundancy) ~= 15.6us at 23.3 TB/s -> memory bound, so fully
// fused single pass. All four 15-tap box passes run on the matrix pipe as
// V_WMMA_F32_16X16X32_F16 against one shared banded 0/1 B-fragment
// (stencil-as-matmul), f16 data / f32 accumulation. Every WMMA stage stores
// its tile TRANSPOSED so the next stage's vertical filter is again a
// contiguous horizontal filter: all LDS traffic is ds_load_b128/ds_store_b128.

typedef __attribute__((ext_vector_type(16))) _Float16 v16h;
typedef __attribute__((ext_vector_type(8)))  _Float16 v8h;
typedef __attribute__((ext_vector_type(4)))  _Float16 v4h;
typedef __attribute__((ext_vector_type(8)))  float    v8f;
typedef __attribute__((ext_vector_type(4)))  float    f4;

#define IMGW 1024
#define IMGH 1024
#define NBATCH 16
#define TILE 128
#define FR   160   // TILE + 2*16 halo (lum frame)
#define R1   146   // TILE + 2*9  (mean / diff^2 region)
#define R2   132   // TILE + 2*2  (combined region)
#define LSTR 168   // all strides multiples of 8 halves -> 16B-aligned chunks
#define PSTR 168
#define QSTR 168
#define MSTR 136

__device__ __forceinline__ float sigmoidf_(float z) {
    return 1.0f / (1.0f + __expf(-z));   // v_exp_f32 + rcp
}

// One 16x16 tile of the 15-tap box filter along the contiguous dimension:
//   acc[m][n] = sum_{k=n..n+14} src[row0+m][col0+k]
// A fragment = two aligned 16B row chunks per lane (CDNA5 ISA 16-bit
// A 16x32 layout: lane<16 holds K=0..7,16..23; lane>=16 holds K=8..15,24..31).
__device__ __forceinline__ v8f hbox15_acc(const _Float16* src, int sstr,
                                          int row0, int col0, int lane,
                                          v16h bband) {
    int m = lane & 15;
    int koff = (lane < 16) ? 0 : 8;
    const _Float16* ap = src + (row0 + m) * sstr + col0 + koff;
    v8h lo = *(const v8h*)ap;          // ds_load_b128
    v8h hi = *(const v8h*)(ap + 16);   // ds_load_b128
    v16h a;
#pragma unroll
    for (int i = 0; i < 8; ++i) { a[i] = lo[i]; a[i + 8] = hi[i]; }
    v8f c = {};
    return __builtin_amdgcn_wmma_f32_16x16x32_f16(false, a, false, bband,
                                                  (short)0, c, false, false);
}

// Store the 16x16 accumulator TRANSPOSED: dst[col0+n][row0+m].
// C layout: acc[i] is (M = i + (lane<16?0:8), N = lane&15), so one lane's 8
// results are contiguous in the transposed row -> single ds_store_b128.
__device__ __forceinline__ void scatterT(v8f acc, _Float16* dst, int dstr,
                                         int row0, int col0, int rowLimit,
                                         int lane) {
    int n = lane & 15;
    int koff = (lane < 16) ? 0 : 8;
    int cT = col0 + n;
    if (cT < rowLimit) {
        v8h o;
#pragma unroll
        for (int i = 0; i < 8; ++i) o[i] = (_Float16)acc[i];
        *(v8h*)(dst + cT * dstr + row0 + koff) = o;   // 16B aligned
    }
}

__global__ __launch_bounds__(256, 1)
void wrd_fused_kernel(const float* __restrict__ img, float* __restrict__ out) {
    // ~193 KB LDS total (< 320 KB/WGP). Allocations sized so every tile
    // overshoot (reads up to row 159 / col 175) stays in-bounds; garbage only
    // reaches stride padding or rows whose outputs are discarded by guards.
    __shared__ _Float16 sLum[FR * LSTR + 64];  // lum, pixel rel [-16,143]^2
    __shared__ _Float16 sP[FR * PSTR + 64];    // hsum(lum)^T -> diff^2 -> combined
    __shared__ _Float16 sQ[R1 * QSTR + 64];    // 225*mean -> hsum(diff^2)^T -> pool tmp
    __shared__ _Float16 sM[R2 * MSTR + 64];    // low-sat mask, pixel rel [-2,129]^2

    const int tid  = threadIdx.x;
    const int wave = tid >> 5;
    const int lane = tid & 31;
    const int n_   = lane & 15;
    const int koff8 = (lane < 16) ? 0 : 8;
    const int gx0 = blockIdx.x * TILE;
    const int gy0 = blockIdx.y * TILE;
    const int b   = blockIdx.z;
    const float* imgb = img + (size_t)b * 3u * IMGH * IMGW;

    // Shared banded 0/1 B fragment, built once, reused by all 4 WMMA stages.
    // B layout matches A: lanes0-15 hold K=0..7,16..23; lanes16-31 hold K=8..15,24..31 at col=lane&15.
    v16h bband;
    {
        int koffb = (lane < 16) ? 0 : 8;
#pragma unroll
        for (int i = 0; i < 16; ++i) {
            int k = (i < 8) ? (koffb + i) : (16 + koffb + (i - 8));
            bband[i] = (k >= n_ && k <= n_ + 14) ? (_Float16)1.0f : (_Float16)0.0f;
        }
    }

    // ---- Stage 0: float4 RGB loads -> lum (zero outside image) + sat mask ----
    // gx0 is a multiple of 128, so each 4-wide x-chunk is wholly in/out of image.
    for (int p = tid; p < FR * (FR / 4); p += 256) {
        int iy = p / (FR / 4), ic = p - iy * (FR / 4);
        int ix = ic * 4;
        int gy = gy0 + iy - 16, gx = gx0 + ix - 16;
        f4 r4 = {0.f, 0.f, 0.f, 0.f}, g4 = r4, b4 = r4;
        bool in = (gy >= 0) & (gy < IMGH) & (gx >= 0) & (gx < IMGW);
        if (in) {
            size_t o = (size_t)gy * IMGW + gx;
            r4 = *(const f4*)(imgb + o);
            g4 = *(const f4*)(imgb + o + (size_t)IMGH * IMGW);
            b4 = *(const f4*)(imgb + o + 2u * (size_t)IMGH * IMGW);
        }
        v4h l4;
#pragma unroll
        for (int e = 0; e < 4; ++e)
            l4[e] = (_Float16)(0.299f * r4[e] + 0.587f * g4[e] + 0.114f * b4[e]);
        *(v4h*)(sLum + iy * LSTR + ix) = l4;           // 8B aligned
        if (iy >= 14 && iy < 14 + R2) {
#pragma unroll
            for (int e = 0; e < 4; ++e) {
                int x = ix + e;
                if (x >= 14 && x < 14 + R2) {
                    float mx = fmaxf(r4[e], fmaxf(g4[e], b4[e]));
                    float mn = fminf(r4[e], fminf(g4[e], b4[e]));
                    float sat = (mx - mn) / (mx + 1e-8f);
                    sM[(iy - 14) * MSTR + (x - 14)] =
                        (_Float16)sigmoidf_((0.3f - sat) * 10.f);
                }
            }
        }
    }
    __syncthreads();

    // ---- Stage 1: H1[y][x'] = 15-sum_x lum  -> stored transposed sP[x'][y] ----
    for (int t = wave; t < 10 * 10; t += 8) {
        int tr = t / 10, tc = t - tr * 10;
        v8f acc = hbox15_acc(sLum, LSTR, tr * 16, tc * 16, lane, bband);
        scatterT(acc, sP, PSTR, tr * 16, tc * 16, R1, lane);
    }
    __syncthreads();

    // ---- Stage 2: 15-sum over y of H1 (= horizontal on sP^T) -> sQ[y'][x'] ----
    // sQ = 225 * local_mean over the 146^2 region.
    for (int t = wave; t < 10 * 10; t += 8) {
        int tr = t / 10, tc = t - tr * 10;
        v8f acc = hbox15_acc(sP, PSTR, tr * 16, tc * 16, lane, bband);
        scatterT(acc, sQ, QSTR, tr * 16, tc * 16, R1, lane);
    }
    __syncthreads();

    // ---- Stage 3: sP[y'][x'] = (lum - mean)^2, zero outside image ----
    // (reference zero-pads the squared term in the conv, so OOB contributes 0)
    for (int p = tid; p < R1 * R1; p += 256) {
        int qy = p / R1, jx = p - qy * R1;
        int gy = gy0 + qy - 9, gx = gx0 + jx - 9;
        float d2 = 0.f;
        if (gy >= 0 && gy < IMGH && gx >= 0 && gx < IMGW) {
            float l = (float)sLum[(qy + 7) * LSTR + (jx + 7)];
            float mean = (float)sQ[qy * QSTR + jx] * (1.0f / 225.0f);
            float d = l - mean;
            d2 = d * d;
        }
        sP[qy * PSTR + jx] = (_Float16)d2;
    }
    __syncthreads();

    // ---- Stage 4: 15-sum_x of diff^2 -> stored transposed sQ[u][y'] ----
    for (int t = wave; t < 10 * 9; t += 8) {
        int tr = t / 9, tc = t - tr * 9;
        v8f acc = hbox15_acc(sP, PSTR, tr * 16, tc * 16, lane, bband);
        scatterT(acc, sQ, QSTR, tr * 16, tc * 16, R2, lane);
    }
    __syncthreads();

    // ---- Stage 5: 15-sum over y (horizontal on sQ^T) -> variance -> combined ----
    for (int t = wave; t < 9 * 9; t += 8) {
        int tr = t / 9, tc = t - tr * 9;
        int row0 = tr * 16, col0 = tc * 16;
        v8f acc = hbox15_acc(sQ, QSTR, row0, col0, lane, bband);
        int vy = col0 + n_;                  // y index in combined region
        if (vy < R2) {
            int ubase = row0 + koff8;        // contiguous u for this lane
            v8h o;
#pragma unroll
            for (int i = 0; i < 8; ++i) {
                int u = ubase + i;
                float S = acc[i];            // 225 * local_var
                float cm = sigmoidf_(S * (50.0f / 225.0f) - 0.5f);
                float l  = (float)sLum[(vy + 14) * LSTR + (u + 14)];
                float bm = sigmoidf_((l - 0.7f) * 10.0f);
                float lm = (float)sM[vy * MSTR + u];
                float comb = fminf(fmaxf(bm + 0.5f * cm * lm, 0.0f), 1.0f);
                int gy = gy0 + vy - 2, gx = gx0 + u - 2;
                if (gy < 0 || gy >= IMGH || gx < 0 || gx >= IMGW) comb = 0.0f;
                o[i] = (_Float16)comb;
            }
            *(v8h*)(sP + vy * PSTR + ubase) = o;   // combined, normal layout
        }
    }
    __syncthreads();

    // ---- Stage 6a: horizontal 5-sum of combined into sQ (132 x 128) ----
    for (int p = tid; p < R2 * TILE; p += 256) {
        int vy = p / TILE, ox = p - vy * TILE;
        float s = 0.f;
#pragma unroll
        for (int k = 0; k < 5; ++k) s += (float)sP[vy * PSTR + ox + k];
        sQ[vy * QSTR + ox] = (_Float16)s;
    }
    __syncthreads();

    // ---- Stage 6b: vertical 5-sum, /25, coalesced f32 stores ----
    float* outb = out + (size_t)b * IMGH * IMGW;
    for (int p = tid; p < TILE * TILE; p += 256) {
        int oy = p / TILE, ox = p - oy * TILE;
        float s = 0.f;
#pragma unroll
        for (int k = 0; k < 5; ++k) s += (float)sQ[(oy + k) * QSTR + ox];
        outb[(size_t)(gy0 + oy) * IMGW + (gx0 + ox)] = s * (1.0f / 25.0f);
    }
}

extern "C" void kernel_launch(void* const* d_in, const int* in_sizes, int n_in,
                              void* d_out, int out_size, void* d_ws, size_t ws_size,
                              hipStream_t stream) {
    (void)in_sizes; (void)n_in; (void)out_size; (void)d_ws; (void)ws_size;
    const float* img = (const float*)d_in[0];
    float* out = (float*)d_out;
    dim3 grid(IMGW / TILE, IMGH / TILE, NBATCH);  // 8 x 8 x 16 workgroups
    dim3 block(256);                              // 8 wave32 waves
    wrd_fused_kernel<<<grid, block, 0, stream>>>(img, out);
}
